// MoeColumnParallelLinear_12000138625733
// MI455X (gfx1250) — matrix operand order repaired and checked by reading, not compiled, not run on hardware
//
#include <hip/hip_runtime.h>

// MoE column-parallel grouped GEMM for MI455X (gfx1250, wave32, WMMA).
// out[t, o] = X[t, :] @ W[e(t), o, :] + bias[e(t), o]
// Strategy: memory-bound problem (~784 MB HBM vs 137 GFLOP) -> use bf16 WMMA
// (16x16x32, f32 accum) so compute stays under the 23.3 TB/s memory roof.

#define T_TOK 131072
#define NE    8
#define DIN   512
#define DOUT  1024

#define BM 128
#define BN 128
#define BK 32
#define LDS_STRIDE 40  // bf16 elems per row: 32 data + 8 pad = 80B, 16B aligned

typedef __attribute__((ext_vector_type(16))) __bf16 bf16x16;
typedef __attribute__((ext_vector_type(8)))  float  f32x8;

union Frag {
  uint4   u[2];
  bf16x16 v;
};

__device__ __forceinline__ unsigned short f32_to_bf16(float f) {
  unsigned u = __builtin_bit_cast(unsigned, f);
  u += 0x7FFFu + ((u >> 16) & 1u);   // round-to-nearest-even
  return (unsigned short)(u >> 16);
}
__device__ __forceinline__ unsigned pack_bf16x2(float lo, float hi) {
  return (unsigned)f32_to_bf16(lo) | ((unsigned)f32_to_bf16(hi) << 16);
}

__global__ __launch_bounds__(256)
void moe_grouped_gemm_wmma(const float* __restrict__ X,
                           const long long* __restrict__ expert_offset,
                           const float* __restrict__ W,
                           const float* __restrict__ Bias,
                           float* __restrict__ out) {
  __shared__ unsigned short ldsA[BM * LDS_STRIDE];
  __shared__ unsigned short ldsB[BN * LDS_STRIDE];

  const int n0 = blockIdx.x * BN;   // output-feature block
  const int m0 = blockIdx.y * BM;   // token block
  const int t  = threadIdx.x;

  // expert_offset is int64 in the reference (jnp.int64)
  long long off[NE + 1];
#pragma unroll
  for (int i = 0; i <= NE; ++i) off[i] = expert_offset[i];

  // searchsorted(right)-1: expert e <=> off[e] <= row < off[e+1]
  int e_lo = 0, e_hi = 0;
#pragma unroll
  for (int e = 0; e < NE; ++e) {
    if (off[e] <= (long long)m0)            e_lo = e;
    if (off[e] <= (long long)(m0 + BM - 1)) e_hi = e;
  }

  if (e_lo == e_hi) {
    // ---------------- fast path: whole 128-token block is one expert --------
    const int e = e_lo;
    const float* Wb = W + (size_t)e * DOUT * DIN;

    const int wave = t >> 5;
    const int lane = t & 31;
    const int half = lane >> 4;   // 0: lanes 0-15, 1: lanes 16-31
    const int l15  = lane & 15;
    const int wm   = wave & 3;    // 4 M-groups x 32 rows
    const int wn   = wave >> 2;   // 2 N-groups x 64 cols

    f32x8 acc[2][4];
#pragma unroll
    for (int i = 0; i < 2; ++i)
#pragma unroll
      for (int j = 0; j < 4; ++j) acc[i][j] = (f32x8)0.0f;

    float4 ax[4], bx[4];  // global staging regs (software pipeline)

    auto loadG = [&](int k0) {
#pragma unroll
      for (int i = 0; i < 4; ++i) {
        int idx = i * 256 + t;       // 1024 float4 per 128x32 f32 tile
        int row = idx >> 3;          // 8 float4 per row
        int c4  = idx & 7;
        ax[i] = *(const float4*)(X  + (size_t)(m0 + row) * DIN + k0 + c4 * 4);
        bx[i] = *(const float4*)(Wb + (size_t)(n0 + row) * DIN + k0 + c4 * 4);
      }
    };
    auto stageLDS = [&]() {
#pragma unroll
      for (int i = 0; i < 4; ++i) {
        int idx = i * 256 + t;
        int row = idx >> 3;
        int c4  = idx & 7;
        uint2 pa, pb;
        pa.x = pack_bf16x2(ax[i].x, ax[i].y);
        pa.y = pack_bf16x2(ax[i].z, ax[i].w);
        pb.x = pack_bf16x2(bx[i].x, bx[i].y);
        pb.y = pack_bf16x2(bx[i].z, bx[i].w);
        *(uint2*)(&ldsA[row * LDS_STRIDE + c4 * 4]) = pa;
        *(uint2*)(&ldsB[row * LDS_STRIDE + c4 * 4]) = pb;
      }
    };

    loadG(0);
    for (int kt = 0; kt < DIN / BK; ++kt) {
      stageLDS();
      __syncthreads();
      if (kt + 1 < DIN / BK) loadG((kt + 1) * BK);  // overlap with WMMA

      // A fragment (16-bit 16x32): lanes 0-15 K={0..7,16..23}, lanes 16-31 +8
      Frag a[2];
#pragma unroll
      for (int mt = 0; mt < 2; ++mt) {
        const unsigned short* p = &ldsA[(wm * 32 + mt * 16 + l15) * LDS_STRIDE];
        a[mt].u[0] = *(const uint4*)(p + half * 8);        // K = half*8 .. +7
        a[mt].u[1] = *(const uint4*)(p + 16 + half * 8);   // K = 16+half*8 ..
      }
      // B fragment (16-bit 32x16): N = lane&15, K = {half*8..+7, 16+half*8..+7}
      Frag b[4];
#pragma unroll
      for (int nt = 0; nt < 4; ++nt) {
        const unsigned short* p = &ldsB[(wn * 64 + nt * 16 + l15) * LDS_STRIDE];
        b[nt].u[0] = *(const uint4*)(p + half * 8);
        b[nt].u[1] = *(const uint4*)(p + 16 + half * 8);
      }

#pragma unroll
      for (int mt = 0; mt < 2; ++mt)
#pragma unroll
        for (int nt = 0; nt < 4; ++nt)
          acc[mt][nt] = __builtin_amdgcn_wmma_f32_16x16x32_bf16(
              false, a[mt].v, false, b[nt].v,
              (short)0, acc[mt][nt], false, false);

      __syncthreads();
    }

    // epilogue: C layout = VGPR r, lanes 0-15 -> M=r, lanes 16-31 -> M=8+r
#pragma unroll
    for (int nt = 0; nt < 4; ++nt) {
      const int o  = n0 + wn * 64 + nt * 16 + l15;
      const float bv = Bias[(size_t)e * DOUT + o];
#pragma unroll
      for (int mt = 0; mt < 2; ++mt) {
        const int mbase = m0 + wm * 32 + mt * 16 + half * 8;
#pragma unroll
        for (int r = 0; r < 8; ++r)
          out[(size_t)(mbase + r) * DOUT + o] = acc[mt][nt][r] + bv;
      }
    }
  } else {
    // ---------------- boundary path: <=7 of 1024 M-blocks straddle experts --
    const int c     = n0 + (t & 127);
    const int rbase = t >> 7;  // two threads per column, 64 rows each
    for (int j = 0; j < BM / 2; ++j) {
      const int m = m0 + rbase + 2 * j;
      int e = 0;
#pragma unroll
      for (int q = 0; q < NE; ++q)
        if (off[q] <= (long long)m) e = q;
      const float* xr = X + (size_t)m * DIN;
      const float* wr = W + ((size_t)e * DOUT + c) * DIN;
      float s = 0.0f;
      for (int k = 0; k < DIN; k += 4) {
        float4 xv = *(const float4*)(xr + k);
        float4 wv = *(const float4*)(wr + k);
        s = fmaf(xv.x, wv.x, s);
        s = fmaf(xv.y, wv.y, s);
        s = fmaf(xv.z, wv.z, s);
        s = fmaf(xv.w, wv.w, s);
      }
      out[(size_t)m * DOUT + c] = s + Bias[(size_t)e * DOUT + c];
    }
  }
}

extern "C" void kernel_launch(void* const* d_in, const int* in_sizes, int n_in,
                              void* d_out, int out_size, void* d_ws, size_t ws_size,
                              hipStream_t stream) {
  (void)in_sizes; (void)n_in; (void)out_size; (void)d_ws; (void)ws_size;
  const float*     X   = (const float*)d_in[0];
  const long long* off = (const long long*)d_in[1];   // int64 per reference
  const float*     W   = (const float*)d_in[2];
  const float*     B   = (const float*)d_in[3];
  float*           out = (float*)d_out;

  dim3 grid(DOUT / BN, T_TOK / BM, 1);   // 8 x 1024 workgroups
  dim3 block(256, 1, 1);                 // 8 wave32 waves
  moe_grouped_gemm_wmma<<<grid, block, 0, stream>>>(X, off, W, B, out);
}
